// VectorQuantizer_62234076119862
// MI455X (gfx1250) — compile-verified, matrix operand order checked by reading, not run on hardware
//
#include <hip/hip_runtime.h>

typedef float v8f  __attribute__((ext_vector_type(8)));
typedef unsigned int v8u __attribute__((ext_vector_type(8)));
typedef __bf16 v16bf __attribute__((ext_vector_type(16)));

#define EMB_DIM     256
#define N_EMBED     1024
#define NROWS       32768              // 32 * 32 * 32
#define TOTAL_ELEMS 8388608            // NROWS * EMB_DIM
#define BPK_DWORDS  131072             // 64 ct * 8 c * 2 half * 32 lane * 4 dw = 512KB

// round-to-nearest-even fp32 -> bf16 (data has no NaN)
__device__ __forceinline__ unsigned int f2bf(float f) {
    unsigned int u = __builtin_bit_cast(unsigned int, f);
    return (u + 0x7fffu + ((u >> 16) & 1u)) >> 16;
}

// ---------------------------------------------------------------------------
// ek2[k] = sum_d codebook[k][d]^2   (fp32)
// ---------------------------------------------------------------------------
__global__ __launch_bounds__(256) void vq_ek2_kernel(const float* __restrict__ cb,
                                                     float* __restrict__ ek2) {
    int k = blockIdx.x * 256 + threadIdx.x;           // 4 blocks
    const float4* row = (const float4*)(cb + (size_t)k * EMB_DIM);
    float s = 0.f;
#pragma unroll
    for (int i = 0; i < EMB_DIM / 4; ++i) {
        float4 v = row[i];
        s += v.x * v.x + v.y * v.y + v.z * v.z + v.w * v.w;
    }
    ek2[k] = s;
}

// ---------------------------------------------------------------------------
// Pre-pack codebook into the per-lane WMMA-B bf16 fragment image.
// Layout (dword index): g = (((ct*8 + c)*2 + half)*32 + L)*4 + within
//   fragment VGPR j = half*4 + within, holds bf16 pair (K0, K0+1) with
//   j<4: K0 = 8*lh + 2*j ; j>=4: K0 = 16 + 8*lh + 2*(j-4)   (lh = L>>4)
//   value = codebook[ct*16 + (L&15)][32*c + K]
// ---------------------------------------------------------------------------
__global__ __launch_bounds__(256) void vq_bpack_kernel(const float* __restrict__ cb,
                                                       unsigned int* __restrict__ bpk) {
    int g = blockIdx.x * 256 + threadIdx.x;           // 512 blocks cover 131072
    int within = g & 3;
    int L      = (g >> 2) & 31;
    int half   = (g >> 7) & 1;
    int c      = (g >> 8) & 7;
    int ct     = g >> 11;
    int j  = half * 4 + within;
    int n  = L & 15;
    int lh = L >> 4;
    int K0 = (j < 4) ? (8 * lh + 2 * j) : (16 + 8 * lh + 2 * (j - 4));
    const float* src = cb + (size_t)(ct * 16 + n) * EMB_DIM + 32 * c + K0;
    bpk[g] = f2bf(src[0]) | (f2bf(src[1]) << 16);
}

// ---------------------------------------------------------------------------
// Main kernel: per 16-row tile, dot(x, e_k) for all 1024 codes with
// v_wmma_f32_16x16x32_bf16 (8 chunks of K=32), argmin of (||e||^2 - 2 dot).
// Block = 256 threads = 8 waves; each wave owns one 16-row tile.
// LDS double-buffered slabs: one barrier per tile, staging latency overlapped.
// ---------------------------------------------------------------------------
__global__ __launch_bounds__(256) void vq_argmin_kernel(const float* __restrict__ enc,
                                                        const unsigned int* __restrict__ bpk,
                                                        const float* __restrict__ ek2g,
                                                        int* __restrict__ idxOut) {
    __shared__ unsigned int ldsB[4096];               // two 8KB bf16 slabs

    const int lane = threadIdx.x & 31;
    const int wave = threadIdx.x >> 5;
    const int nn   = lane & 15;        // M (A) / N (B,C) within tile
    const int lh   = lane >> 4;        // K-half select

    const int row0 = blockIdx.x * 128 + wave * 16;    // 256 blocks cover 32768 rows
    const int b    = row0 >> 10;
    const int hw0  = row0 & 1023;

    // flat[n][d] = enc[b*262144 + d*1024 + hw],  n = b*1024 + hw
    const float* aPtr = enc + (size_t)b * 262144 + hw0 + nn;

    // Preload + convert A fragments for all 8 K-chunks (16-bit A 16x32 layout):
    // VGPR j<4: K = 8*lh + 2j,+1 ; j>=4: K = 16 + 8*lh + 2(j-4),+1
    v16bf afrag[8];
#pragma unroll
    for (int c = 0; c < 8; ++c) {
        v8u w;
#pragma unroll
        for (int j = 0; j < 8; ++j) {
            int K0 = (j < 4) ? (8 * lh + 2 * j) : (16 + 8 * lh + 2 * (j - 4));
            int d  = 32 * c + K0;
            float f0 = aPtr[(size_t)d * 1024];
            float f1 = aPtr[(size_t)(d + 1) * 1024];
            w[j] = f2bf(f0) | (f2bf(f1) << 16);
        }
        afrag[c] = __builtin_bit_cast(v16bf, w);
    }

    float best[8];
    int   bidx[8];
#pragma unroll
    for (int r = 0; r < 8; ++r) { best[r] = 3.4e38f; bidx[r] = 0; }

    // Prologue: stage slab 0 into buffer 0 (straight contiguous 8KB copy)
    {
        const uint4* src = (const uint4*)bpk;
        uint4* dst = (uint4*)ldsB;
        dst[threadIdx.x]       = src[threadIdx.x];
        dst[threadIdx.x + 256] = src[threadIdx.x + 256];
    }
    __syncthreads();

    for (int ct = 0; ct < 64; ++ct) {                 // 64 column tiles of 16 codes
        // Issue next-slab global loads first; latency hidden by compute below.
        uint4 p0 = {}, p1 = {};
        if (ct < 63) {
            const uint4* src = (const uint4*)(bpk + (ct + 1) * 2048);
            p0 = src[threadIdx.x];
            p1 = src[threadIdx.x + 256];
        }

        float e2 = ek2g[ct * 16 + nn];

        v8f acc = {0.f, 0.f, 0.f, 0.f, 0.f, 0.f, 0.f, 0.f};
        const uint4* lb = (const uint4*)(ldsB + (ct & 1) * 2048);
#pragma unroll
        for (int c = 0; c < 8; ++c) {
            uint4 lo = lb[c * 64 + lane];             // fragment VGPRs 0..3
            uint4 hi = lb[c * 64 + 32 + lane];        // fragment VGPRs 4..7
            v8u w;
            w[0] = lo.x; w[1] = lo.y; w[2] = lo.z; w[3] = lo.w;
            w[4] = hi.x; w[5] = hi.y; w[6] = hi.z; w[7] = hi.w;
            v16bf bv = __builtin_bit_cast(v16bf, w);
            acc = __builtin_amdgcn_wmma_f32_16x16x32_bf16(
                false, afrag[c], false, bv, (short)0, acc, false, false);
        }

        // C layout: lanes 0-15 hold (M=r, N=nn); lanes 16-31 hold (M=8+r, N=nn)
        const int cand = ct * 16 + nn;                // ascending -> first-min kept
#pragma unroll
        for (int r = 0; r < 8; ++r) {
            float v = e2 - 2.0f * acc[r];
            if (v < best[r]) { best[r] = v; bidx[r] = cand; }
        }

        // Publish next slab into the other buffer; single barrier per tile.
        // (Barrier at end of iter ct-1 guarantees buffer (ct+1)&1 is free.)
        if (ct < 63) {
            uint4* dst = (uint4*)(ldsB + ((ct + 1) & 1) * 2048);
            dst[threadIdx.x]       = p0;
            dst[threadIdx.x + 256] = p1;
            __syncthreads();
        }
    }

    // Min-reduce (value, index) across the 16 lanes holding the same row.
#pragma unroll
    for (int r = 0; r < 8; ++r) {
#pragma unroll
        for (int off = 1; off < 16; off <<= 1) {
            float ov = __shfl_xor(best[r], off, 32);
            int   oi = __shfl_xor(bidx[r], off, 32);
            if (ov < best[r] || (ov == best[r] && oi < bidx[r])) {
                best[r] = ov; bidx[r] = oi;
            }
        }
    }
    if (nn == 0) {                                    // lanes 0 (M=0..7) and 16 (M=8..15)
#pragma unroll
        for (int r = 0; r < 8; ++r)
            idxOut[row0 + lh * 8 + r] = bidx[r];
    }
}

// ---------------------------------------------------------------------------
// Deterministic fp32 MSE partial sums: sum (enc[i] - codebook[idx[n]][d])^2
// ---------------------------------------------------------------------------
__global__ __launch_bounds__(256) void vq_mse_partial_kernel(const float* __restrict__ enc,
                                                             const float* __restrict__ cb,
                                                             const int* __restrict__ idx,
                                                             float* __restrict__ partial) {
    __shared__ float red[256];
    float s = 0.f;
    for (int i = blockIdx.x * 256 + threadIdx.x; i < TOTAL_ELEMS; i += 256 * 256) {
        int b   = i >> 18;             // 262144 elems per batch
        int rem = i & 262143;
        int d   = rem >> 10;
        int hw  = rem & 1023;
        int n   = (b << 10) + hw;
        float diff = enc[i] - cb[(size_t)idx[n] * EMB_DIM + d];
        s += diff * diff;
    }
    red[threadIdx.x] = s;
    __syncthreads();
    for (int o = 128; o > 0; o >>= 1) {
        if (threadIdx.x < (unsigned)o) red[threadIdx.x] += red[threadIdx.x + o];
        __syncthreads();
    }
    if (threadIdx.x == 0) partial[blockIdx.x] = red[0];
}

__global__ __launch_bounds__(256) void vq_finalize_kernel(const float* __restrict__ partial,
                                                          float* __restrict__ outLoss) {
    __shared__ float red[256];
    red[threadIdx.x] = partial[threadIdx.x];
    __syncthreads();
    for (int o = 128; o > 0; o >>= 1) {
        if (threadIdx.x < (unsigned)o) red[threadIdx.x] += red[threadIdx.x + o];
        __syncthreads();
    }
    if (threadIdx.x == 0) {
        float mse = red[0] * (1.0f / (float)TOTAL_ELEMS);
        outLoss[0] = 1.25f * mse;   // loss = codebook + 0.25 * commitment
        outLoss[1] = mse;           // codebook_loss
        outLoss[2] = mse;           // commitment_loss (identical in value)
    }
}

// ---------------------------------------------------------------------------
extern "C" void kernel_launch(void* const* d_in, const int* in_sizes, int n_in,
                              void* d_out, int out_size, void* d_ws, size_t ws_size,
                              hipStream_t stream) {
    const float* enc = (const float*)d_in[0];   // [32,256,32,32]
    const float* cb  = (const float*)d_in[1];   // [1024,256]
    float* out = (float*)d_out;                 // 8388608 + 3 floats

    float*        ek2     = (float*)d_ws;                              // 4KB
    int*          idx     = (int*)((char*)d_ws + 4096);                // 128KB
    float*        partial = (float*)((char*)d_ws + 4096 + 131072);     // 1KB
    unsigned int* bpk     = (unsigned int*)((char*)d_ws + 262144);     // 512KB

    // Faithful-bug output: out == encoderout.
    hipMemcpyAsync(out, enc, (size_t)TOTAL_ELEMS * sizeof(float),
                   hipMemcpyDeviceToDevice, stream);

    vq_ek2_kernel<<<N_EMBED / 256, 256, 0, stream>>>(cb, ek2);
    vq_bpack_kernel<<<BPK_DWORDS / 256, 256, 0, stream>>>(cb, bpk);
    vq_argmin_kernel<<<NROWS / 128, 256, 0, stream>>>(enc, bpk, ek2, idx);
    vq_mse_partial_kernel<<<256, 256, 0, stream>>>(enc, cb, idx, partial);
    vq_finalize_kernel<<<1, 256, 0, stream>>>(partial, out + TOTAL_ELEMS);
}
